// MotionNetwork_53334903881937
// MI455X (gfx1250) — compile-verified, matrix operand order checked by reading, not run on hardware
//
#include <hip/hip_runtime.h>
#include <math.h>

// ---------------------------------------------------------------------------
// MotionNetwork for MI455X (gfx1250, wave32, WMMA).
// Compute-bound (~27.5 GFLOP vs ~90MB HBM): run all per-point MLPs on
// v_wmma_f32_16x16x32_f16, one 16-point tile per wave, activations staged in
// per-wave LDS in the ISA A-fragment-friendly layout.
// ---------------------------------------------------------------------------

typedef __attribute__((ext_vector_type(16))) _Float16 v16h;
typedef __attribute__((ext_vector_type(8)))  _Float16 v8h;
typedef __attribute__((ext_vector_type(8)))  float    v8f;

#define L_LEVELS 12
#define N_FRAGS  36

// Fragment base offsets (each fragment = 32 lanes * 8 dwords = 1KB)
#define FR_AUD1 0   // 2kt x 2nt
#define FR_AUD2 4   // 1kt x 2nt
#define FR_EYE1 6   // 2kt x 1nt
#define FR_EYE2 8   // 1kt x 1nt
#define FR_SIG1 9   // 3kt x 4nt
#define FR_SIG2 21  // 2kt x 4nt
#define FR_SIG3 29  // 2kt x 1nt
#define FR_TMP1 31  // 2kt x 2nt
#define FR_TMP2 35  // 1kt x 1nt

struct GridConst { float scale[L_LEVELS]; int off[L_LEVELS]; int stride[L_LEVELS]; };

struct MainArgs {
    const float* x;
    const float* tab0; const float* tab1; const float* tab2;
    const unsigned int* pack;   // packed f16 B-fragments in ws
    const float* encA;          // 32 f32 (audio feature, per call)
    const float* encE;          // 6  f32 (expression feature, per call)
    float* out;                 // [npts][19]
    int npts;
    GridConst g;
};

struct PrepArgs {
    const float* a; const float* e;
    const float* w1; const float* b1;
    const float* w2; const float* b2;
    const float* w3; const float* b3;
    const float* cw[5]; const float* cb[5];
    const float* aw; const float* ab;
    const float* exp1; const float* exp2;
    float* encA; float* encE;
    unsigned int* pack;
    const float* Wm[9];          // weight matrices to pack
    int N[9]; int K[9]; int NT[9];
    int FB[10];                  // fragment base per layer (+sentinel 36)
};

static __device__ __forceinline__ float leakyf(float x) { return x >= 0.f ? x : 0.02f * x; }

static __device__ __forceinline__ void wsync() {
    asm volatile("s_wait_dscnt 0x0" ::: "memory");
    __builtin_amdgcn_wave_barrier();
}

// A-fragment: 16x32 f16, lane holds row M=lane&15; halves 0..7 -> K=kb..kb+7,
// halves 8..15 -> K=kb+16..kb+23 where kb = kt*32 + (lane>>4)*8.
static __device__ __forceinline__ v16h ldA(const _Float16* rowp, int kb) {
    union { v16h v; v8h h[2]; } u;
    u.h[0] = *(const v8h*)(rowp + kb);
    u.h[1] = *(const v8h*)(rowp + kb + 16);
    return u.v;
}

// B-fragment from workspace: frag*1KB + lane*32B (two 16B loads, L2-resident).
static __device__ __forceinline__ v16h ldB(const unsigned int* pack, int frag, int lane) {
    const unsigned int* p = pack + frag * 256 + lane * 8;
    union { v16h v; v8h h[2]; } u;
    u.h[0] = *(const v8h*)(p);
    u.h[1] = *(const v8h*)(p + 4);
    return u.v;
}

static __device__ __forceinline__ v8f wf(v16h a, v16h b, v8f c) {
    return __builtin_amdgcn_wmma_f32_16x16x32_f16(false, a, false, b, (short)0, c, false, false);
}

// Store D tile (lane holds col N=lane&15, rows M=v+8*(lane>>4)) to LDS f16.
// base must point at lds[8*(lane>>4)][lane&15]; col0 selects the N-tile.
static __device__ __forceinline__ void st16(_Float16* base, int col0, v8f c, bool relu) {
#pragma unroll
    for (int v = 0; v < 8; ++v) {
        float x = c[v];
        if (relu) x = fmaxf(x, 0.f);
        base[v * 96 + col0] = (_Float16)x;
    }
}

// ---------------------------------------------------------------------------
// Prep kernel: audio encoder + expression MLP + weight packing (1 block).
// ---------------------------------------------------------------------------
__global__ __launch_bounds__(256) void motion_prep(PrepArgs P) {
    __shared__ float sh1[8 * 256];
    __shared__ float sh2[8 * 128];
    __shared__ float senc[8 * 32];
    __shared__ float scv[2][32 * 8];
    __shared__ float satt[24];
    __shared__ float sexp[16];
    const int tid = threadIdx.x;

    // afc1: (8,512)->(8,256), leaky
#pragma unroll 1
    for (int i = 0; i < 8; ++i) {
        int o = tid + i * 256; int s = o >> 8, c = o & 255;
        float acc = P.b1[c];
        for (int k = 0; k < 512; ++k) acc += P.a[s * 512 + k] * P.w1[c * 512 + k];
        sh1[s * 256 + c] = leakyf(acc);
    }
    __syncthreads();
    // afc2: ->(8,128), leaky
#pragma unroll 1
    for (int i = 0; i < 4; ++i) {
        int o = tid + i * 256; int s = o >> 7, c = o & 127;
        float acc = P.b2[c];
        for (int k = 0; k < 256; ++k) acc += sh1[s * 256 + k] * P.w2[c * 256 + k];
        sh2[s * 128 + c] = leakyf(acc);
    }
    __syncthreads();
    // afc3: ->(8,32), no act
    {
        int s = tid >> 5, c = tid & 31;
        float acc = P.b3[c];
        for (int k = 0; k < 128; ++k) acc += sh2[s * 128 + k] * P.w3[c * 128 + k];
        senc[s * 32 + c] = acc;
    }
    __syncthreads();
    // conv input: (1,32,8), y[c][t] = enc[t][c]
    { int c = tid >> 3, t = tid & 7; scv[0][c * 8 + t] = senc[t * 32 + c]; }
    __syncthreads();
    const int chans[6] = {32, 16, 8, 4, 2, 1};
    int cur = 0;
#pragma unroll 1
    for (int i = 0; i < 5; ++i) {
        int Cin = chans[i], Cout = chans[i + 1];
        if (tid < Cout * 8) {
            int co = tid >> 3, t = tid & 7;
            float acc = P.cb[i][co];
            for (int ci = 0; ci < Cin; ++ci)
                for (int dk = 0; dk < 3; ++dk) {
                    int tt = t + dk - 1;
                    if (tt >= 0 && tt < 8)
                        acc += scv[cur][ci * 8 + tt] * P.cw[i][(co * Cin + ci) * 3 + dk];
                }
            scv[cur ^ 1][co * 8 + t] = leakyf(acc);
        }
        __syncthreads();
        cur ^= 1;
    }
    // attention logits + softmax + enc_a
    if (tid < 8) {
        float acc = P.ab[tid];
        for (int t = 0; t < 8; ++t) acc += scv[cur][t] * P.aw[tid * 8 + t];
        satt[tid] = acc;
    }
    __syncthreads();
    if (tid == 0) {
        float m = satt[0];
        for (int s = 1; s < 8; ++s) m = fmaxf(m, satt[s]);
        float sum = 0.f;
        for (int s = 0; s < 8; ++s) { float ev = __expf(satt[s] - m); satt[8 + s] = ev; sum += ev; }
        for (int s = 0; s < 8; ++s) satt[8 + s] /= sum;
    }
    __syncthreads();
    if (tid < 32) {
        float acc = 0.f;
        for (int s = 0; s < 8; ++s) acc += satt[8 + s] * senc[s * 32 + tid];
        P.encA[tid] = acc;
    }
    // expression MLP: 5->16 relu ->5, concat e[5]
    if (tid < 16) {
        float acc = 0.f;
        for (int k = 0; k < 5; ++k) acc += P.e[k] * P.exp1[tid * 5 + k];
        sexp[tid] = fmaxf(acc, 0.f);
    }
    __syncthreads();
    if (tid < 5) {
        float acc = 0.f;
        for (int k = 0; k < 16; ++k) acc += sexp[k] * P.exp2[tid * 16 + k];
        P.encE[tid] = acc;
    }
    if (tid == 5) P.encE[5] = P.e[5];

    // ---- pack MLP weights into f16 WMMA B-fragments (B[k][n] = W[n][k]) ----
#pragma unroll 1
    for (int u = tid; u < N_FRAGS * 256; u += 256) {
        int frag = u >> 8, dw = u & 255;
        int l = 0;
        while (l < 8 && frag >= P.FB[l + 1]) ++l;
        int local = frag - P.FB[l];
        int NT = P.NT[l];
        int kt = local / NT, nt = local % NT;
        int lane = dw >> 3, d = dw & 7;
        int n = nt * 16 + (lane & 15);
        int hi = lane >> 4;
        int k0 = kt * 32 + ((d < 4) ? 0 : 16) + hi * 8 + ((2 * d) & 7);
        const float* W = P.Wm[l];
        int Nn = P.N[l], Kk = P.K[l];
        float v0 = (n < Nn && k0     < Kk) ? W[n * Kk + k0]     : 0.f;
        float v1 = (n < Nn && k0 + 1 < Kk) ? W[n * Kk + k0 + 1] : 0.f;
        union { _Float16 h[2]; unsigned int u32; } pk;
        pk.h[0] = (_Float16)v0; pk.h[1] = (_Float16)v1;
        P.pack[frag * 256 + dw] = pk.u32;
    }
}

// ---------------------------------------------------------------------------
// Main kernel: one 16-point tile per wave, grid-stride over tiles.
// ---------------------------------------------------------------------------
__global__ __launch_bounds__(256) void motion_main(MainArgs A) {
    __shared__ _Float16 sact[8][16][96];   // enc_x(36) | enc_w(32) | enc_e(6) | pad
    __shared__ _Float16 shid[8][16][96];   // hidden activations
    const int tid  = threadIdx.x;
    const int w    = tid >> 5;
    const int lane = tid & 31;
    const int hi   = lane >> 4;
    const int row  = lane & 15;            // M row for A loads
    const int n    = lane & 15;            // N col for B/D
    _Float16* actRow = &sact[w][row][0];
    _Float16* hidRow = &shid[w][row][0];
    _Float16* actD   = &sact[w][hi * 8][n];
    _Float16* hidD   = &shid[w][hi * 8][n];
    const float* tabs[3] = {A.tab0, A.tab1, A.tab2};

    const int ntiles = (A.npts + 15) >> 4;
    for (int tile = blockIdx.x * 8 + w; tile < ntiles; tile += gridDim.x * 8) {
        const int base = tile << 4;

        // zero pad region (cols 36..95) of sact
#pragma unroll 1
        for (int i = lane; i < 16 * 30; i += 32) {
            int r = i / 30, d = i % 30;
            *((unsigned int*)&sact[w][r][36] + d) = 0u;
        }

        // ---- grid encode: lane handles point 'row', feats f = hi + 2j ----
        {
            int P = base + row; if (P >= A.npts) P = A.npts - 1;
            const float* xp = A.x + (size_t)3 * P;
            float x0 = xp[0], x1 = xp[1], x2 = xp[2];
#pragma unroll 1
            for (int j = 0; j < 18; ++j) {
                int f = hi + 2 * j;
                int plane = f / 12, level = f % 12;
                float u0 = (plane == 1) ? x1 : x0;
                float u1 = (plane == 0) ? x1 : x2;
                u0 = fminf(fmaxf((u0 + 0.15f) * (1.f / 0.3f), 0.f), 1.f);
                u1 = fminf(fmaxf((u1 + 0.15f) * (1.f / 0.3f), 0.f), 1.f);
                float sc = A.g.scale[level];
                float p0 = u0 * sc + 0.5f, p1 = u1 * sc + 0.5f;
                float g0 = floorf(p0), g1 = floorf(p1);
                float fx = p0 - g0, fy = p1 - g1;
                int st = A.g.stride[level];
                int i00 = A.g.off[level] + (int)g0 * st + (int)g1;
                const float* t = tabs[plane];
                float t00 = t[i00], t10 = t[i00 + st], t01 = t[i00 + 1], t11 = t[i00 + st + 1];
                float feat = t00 * (1.f - fx) * (1.f - fy) + t10 * fx * (1.f - fy)
                           + t01 * (1.f - fx) * fy + t11 * fx * fy;
                sact[w][row][f] = (_Float16)feat;
            }
        }
        wsync();

        // ---- aud1: 36->32 relu ----
        {
            v8f c0 = {}, c1 = {};
#pragma unroll
            for (int kt = 0; kt < 2; ++kt) {
                v16h a = ldA(actRow, kt * 32 + hi * 8);
                c0 = wf(a, ldB(A.pack, FR_AUD1 + kt * 2 + 0, lane), c0);
                c1 = wf(a, ldB(A.pack, FR_AUD1 + kt * 2 + 1, lane), c1);
            }
            st16(hidD, 0, c0, true);
            st16(hidD, 16, c1, true);
        }
        // ---- eye1: 36->16 relu (+ zero cols 48..63 for padded K of eye2) ----
        {
            v8f c0 = {};
#pragma unroll
            for (int kt = 0; kt < 2; ++kt) {
                v16h a = ldA(actRow, kt * 32 + hi * 8);
                c0 = wf(a, ldB(A.pack, FR_EYE1 + kt, lane), c0);
            }
            st16(hidD, 32, c0, true);
#pragma unroll
            for (int v = 0; v < 8; ++v) hidD[v * 96 + 48] = (_Float16)0.f;
        }
        // ---- tmp1: 36->32 relu ----
        {
            v8f c0 = {}, c1 = {};
#pragma unroll
            for (int kt = 0; kt < 2; ++kt) {
                v16h a = ldA(actRow, kt * 32 + hi * 8);
                c0 = wf(a, ldB(A.pack, FR_TMP1 + kt * 2 + 0, lane), c0);
                c1 = wf(a, ldB(A.pack, FR_TMP1 + kt * 2 + 1, lane), c1);
            }
            st16(hidD, 64, c0, true);
            st16(hidD, 80, c1, true);
        }
        wsync();

        // ---- aud2: 32->32 (raw), amb_aud, enc_w = enc_a * att ----
        {
            v16h a = ldA(hidRow, hi * 8);
            v8f c0 = {}, c1 = {};
            c0 = wf(a, ldB(A.pack, FR_AUD2 + 0, lane), c0);
            c1 = wf(a, ldB(A.pack, FR_AUD2 + 1, lane), c1);
            float ea0 = A.encA[n], ea1 = A.encA[16 + n];
#pragma unroll
            for (int v = 0; v < 8; ++v) {
                int pi = base + v + 8 * hi;
                float a0 = c0[v], a1 = c1[v];
                float sq = a0 * a0 + a1 * a1;
                sq += __shfl_xor(sq, 1); sq += __shfl_xor(sq, 2);
                sq += __shfl_xor(sq, 4); sq += __shfl_xor(sq, 8);
                if (n == 0 && pi < A.npts) A.out[(size_t)pi * 19 + 11] = sqrtf(sq);
                actD[v * 96 + 36] = (_Float16)(a0 * ea0);
                actD[v * 96 + 52] = (_Float16)(a1 * ea1);
            }
        }
        // ---- eye2: 16->6, outer relu, amb_eye, enc_e broadcast * att ----
        {
            v16h a = ldA(hidRow, 32 + hi * 8);
            v8f c0 = {};
            c0 = wf(a, ldB(A.pack, FR_EYE2, lane), c0);
            float ee = (n < 6) ? A.encE[n] : 0.f;
#pragma unroll
            for (int v = 0; v < 8; ++v) {
                int pi = base + v + 8 * hi;
                float att = fmaxf(c0[v], 0.f);
                float sq = att * att;
                sq += __shfl_xor(sq, 1); sq += __shfl_xor(sq, 2);
                sq += __shfl_xor(sq, 4); sq += __shfl_xor(sq, 8);
                if (n == 0 && pi < A.npts) A.out[(size_t)pi * 19 + 12] = sqrtf(sq);
                if (n < 6) actD[v * 96 + 68] = (_Float16)(ee * att);
            }
        }
        // ---- tmp2: 32->6 (raw) -> temp_offset / temp_scale ----
        {
            v16h a = ldA(hidRow, 64 + hi * 8);
            v8f c0 = {};
            c0 = wf(a, ldB(A.pack, FR_TMP2, lane), c0);
#pragma unroll
            for (int v = 0; v < 8; ++v) {
                int pi = base + v + 8 * hi;
                if (n < 6 && pi < A.npts) {
                    float t = c0[v];
                    float o = (n < 3) ? t * 0.001f
                                      : (1.f / (1.f + __expf(-t)) * 0.5f + 0.75f);
                    A.out[(size_t)pi * 19 + 13 + n] = o;
                }
            }
        }
        wsync();

        // ---- sig1: 74(->96)->64 relu ----
        {
            v8f c0 = {}, c1 = {}, c2 = {}, c3 = {};
#pragma unroll
            for (int kt = 0; kt < 3; ++kt) {
                v16h a = ldA(actRow, kt * 32 + hi * 8);
                c0 = wf(a, ldB(A.pack, FR_SIG1 + kt * 4 + 0, lane), c0);
                c1 = wf(a, ldB(A.pack, FR_SIG1 + kt * 4 + 1, lane), c1);
                c2 = wf(a, ldB(A.pack, FR_SIG1 + kt * 4 + 2, lane), c2);
                c3 = wf(a, ldB(A.pack, FR_SIG1 + kt * 4 + 3, lane), c3);
            }
            st16(hidD, 0, c0, true);  st16(hidD, 16, c1, true);
            st16(hidD, 32, c2, true); st16(hidD, 48, c3, true);
        }
        wsync();
        // ---- sig2: 64->64 relu (in place; all A loads precede stores) ----
        {
            v8f c0 = {}, c1 = {}, c2 = {}, c3 = {};
#pragma unroll
            for (int kt = 0; kt < 2; ++kt) {
                v16h a = ldA(hidRow, kt * 32 + hi * 8);
                c0 = wf(a, ldB(A.pack, FR_SIG2 + kt * 4 + 0, lane), c0);
                c1 = wf(a, ldB(A.pack, FR_SIG2 + kt * 4 + 1, lane), c1);
                c2 = wf(a, ldB(A.pack, FR_SIG2 + kt * 4 + 2, lane), c2);
                c3 = wf(a, ldB(A.pack, FR_SIG2 + kt * 4 + 3, lane), c3);
            }
            st16(hidD, 0, c0, true);  st16(hidD, 16, c1, true);
            st16(hidD, 32, c2, true); st16(hidD, 48, c3, true);
        }
        wsync();
        // ---- sig3: 64->11 (raw) -> d_xyz*0.01, d_rot, d_opa, d_scale ----
        {
            v8f c0 = {};
#pragma unroll
            for (int kt = 0; kt < 2; ++kt) {
                v16h a = ldA(hidRow, kt * 32 + hi * 8);
                c0 = wf(a, ldB(A.pack, FR_SIG3 + kt, lane), c0);
            }
#pragma unroll
            for (int v = 0; v < 8; ++v) {
                int pi = base + v + 8 * hi;
                if (n < 11 && pi < A.npts) {
                    float h = c0[v];
                    A.out[(size_t)pi * 19 + n] = (n < 3) ? h * 0.01f : h;
                }
            }
        }
        wsync();
    }
}

// ---------------------------------------------------------------------------
extern "C" void kernel_launch(void* const* d_in, const int* in_sizes, int n_in,
                              void* d_out, int out_size, void* d_ws, size_t ws_size,
                              hipStream_t stream) {
    // Input order (setup_inputs, params flattened in insertion order):
    // 0:x 1:a 2:e 3:table_xy 4:table_yz 5:table_xz 6..11:afc{1,2,3}_{w,b}
    // 12..21:cw0,cb0..cw4,cb4 22:aw 23:ab 24:aud1 25:aud2 26:eye1 27:eye2
    // 28:exp1 29:exp2 30:sig1 31:sig2 32:sig3 33:tmp1 34:tmp2
    const int npts = in_sizes[0] / 3;

    unsigned int* pack = (unsigned int*)d_ws;              // 36 KB
    float* encA = (float*)((char*)d_ws + N_FRAGS * 1024);  // 32 f32
    float* encE = encA + 32;                               // 6 f32

    PrepArgs P;
    P.a  = (const float*)d_in[1];  P.e  = (const float*)d_in[2];
    P.w1 = (const float*)d_in[6];  P.b1 = (const float*)d_in[7];
    P.w2 = (const float*)d_in[8];  P.b2 = (const float*)d_in[9];
    P.w3 = (const float*)d_in[10]; P.b3 = (const float*)d_in[11];
    for (int i = 0; i < 5; ++i) {
        P.cw[i] = (const float*)d_in[12 + 2 * i];
        P.cb[i] = (const float*)d_in[13 + 2 * i];
    }
    P.aw = (const float*)d_in[22]; P.ab = (const float*)d_in[23];
    P.exp1 = (const float*)d_in[28]; P.exp2 = (const float*)d_in[29];
    P.encA = encA; P.encE = encE; P.pack = pack;
    // layer order: aud1 aud2 eye1 eye2 sig1 sig2 sig3 tmp1 tmp2
    const int widx[9] = {24, 25, 26, 27, 30, 31, 32, 33, 34};
    const int Narr[9] = {32, 32, 16, 6, 64, 64, 11, 32, 6};
    const int Karr[9] = {36, 32, 36, 16, 74, 64, 64, 36, 32};
    const int NTar[9] = {2, 2, 1, 1, 4, 4, 1, 2, 1};
    const int FBar[10] = {0, 4, 6, 8, 9, 21, 29, 31, 35, 36};
    for (int i = 0; i < 9; ++i) {
        P.Wm[i] = (const float*)d_in[widx[i]];
        P.N[i] = Narr[i]; P.K[i] = Karr[i]; P.NT[i] = NTar[i]; P.FB[i] = FBar[i];
    }
    P.FB[9] = FBar[9];

    motion_prep<<<1, 256, 0, stream>>>(P);

    MainArgs M;
    M.x = (const float*)d_in[0];
    M.tab0 = (const float*)d_in[3];
    M.tab1 = (const float*)d_in[4];
    M.tab2 = (const float*)d_in[5];
    M.pack = pack; M.encA = encA; M.encE = encE;
    M.out = (float*)d_out; M.npts = npts;
    {   // grid constants (match reference: BOUND=0.15, L=12, BASE=16)
        double G = pow(256.0 * 0.15 / 16.0, 1.0 / 11.0);
        long off = 0;
        for (int l = 0; l < L_LEVELS; ++l) {
            double s = 16.0 * pow(G, (double)l) - 1.0;
            int res = (int)ceil(s) + 1;
            int st = res + 1;
            M.g.scale[l] = (float)s;
            M.g.stride[l] = st;
            M.g.off[l] = (int)off;
            off += (long)st * st;
        }
    }
    motion_main<<<2048, 256, 0, stream>>>(M);
}
